// UNet3DModel_2224793059401
// MI455X (gfx1250) — matrix operand order repaired and checked by reading, not compiled, not run on hardware
//
#include <hip/hip_runtime.h>
#include <hip/hip_bf16.h>

#define Nn   131072
#define Ee   917504
#define Cc   64
#define COUT 64
#define EMBc 256
#define Bb   8
#define NET  7
#define NNT  7
#define NC   71          // Cc + NNT
#define NCP  72          // padded per-edge-type block
#define KK   512         // padded K (NET*NCP = 504 -> 512)
#define CPG  2           // channels per group (64 ch / 32 groups)
#define EPSf 1e-5f

typedef __attribute__((ext_vector_type(16))) __bf16 v16bf;
typedef __attribute__((ext_vector_type(8)))  float  v8f;

__device__ __forceinline__ unsigned short f2bfu(float f) {
    union { float f; unsigned u; } in; in.f = f;
    unsigned u = in.u;
    return (unsigned short)((u + 0x7FFFu + ((u >> 16) & 1u)) >> 16);   // RNE
}

__device__ __forceinline__ float silu(float v) {
    return v / (1.0f + __expf(-v));
}

// packed bf16x2 atomic add -> global_atomic_pk_add_bf16
__device__ __forceinline__ void atomPkAddBf16(unsigned short* p, unsigned payload) {
    __hip_bfloat162_raw r;
    r.x = (unsigned short)(payload & 0xFFFFu);
    r.y = (unsigned short)(payload >> 16);
    __hip_bfloat162 v(r);
    unsafeAtomicAdd((__hip_bfloat162*)p, v);
}

// ---------------- per-batch sum / sumsq / count (batch_id is sorted) -------
__global__ void k_stats(const float* __restrict__ src, const int* __restrict__ bid,
                        float* __restrict__ sums, float* __restrict__ sumsq,
                        float* __restrict__ counts, int rows_per_block) {
    int c  = threadIdx.x;                 // 64 threads = channel
    int r0 = blockIdx.x * rows_per_block;
    int r1 = r0 + rows_per_block; if (r1 > Nn) r1 = Nn;
    if (r0 >= Nn) return;
    int curb = bid[r0];
    float s = 0.f, q = 0.f; int cnt = 0;
    for (int r = r0; r < r1; ++r) {
        int b = bid[r];
        if (b != curb) {
            atomicAdd(&sums[curb * Cc + c], s);
            atomicAdd(&sumsq[curb * Cc + c], q);
            if (c == 0) atomicAdd(&counts[curb], (float)cnt);
            s = 0.f; q = 0.f; cnt = 0; curb = b;
        }
        float v = src[(size_t)r * Cc + c];
        s += v; q += v * v; ++cnt;
    }
    atomicAdd(&sums[curb * Cc + c], s);
    atomicAdd(&sumsq[curb * Cc + c], q);
    if (c == 0) atomicAdd(&counts[curb], (float)cnt);
}

// ------- group-norm (32 groups, cpg=2) + SiLU; writes packed bf16 pairs ----
// one thread per (node, channel-pair); a pair IS one group, so mean/var once.
__global__ void k_gn_silu(const float* __restrict__ src, const int* __restrict__ bid,
                          const float* __restrict__ sums, const float* __restrict__ sumsq,
                          const float* __restrict__ counts,
                          const float* __restrict__ gamma, const float* __restrict__ beta,
                          unsigned* __restrict__ dst) {
    size_t i = (size_t)blockIdx.x * blockDim.x + threadIdx.x;
    if (i >= (size_t)Nn * (Cc / 2)) return;
    int n  = (int)(i >> 5);
    int cp = (int)(i & 31);
    int c0 = cp * 2;
    int b  = bid[n];
    float cntf  = counts[b];
    float denom = cntf * CPG + EPSf;
    float sg = sums[b * Cc + c0] + sums[b * Cc + c0 + 1];
    float qg = sumsq[b * Cc + c0] + sumsq[b * Cc + c0 + 1];
    float mean = sg / denom;
    float var  = (qg - 2.f * mean * sg + cntf * CPG * mean * mean) / denom;
    float inv  = rsqrtf(var + EPSf);
    float v0 = (src[(size_t)n * Cc + c0]     - mean) * inv * gamma[c0]     + beta[c0];
    float v1 = (src[(size_t)n * Cc + c0 + 1] - mean) * inv * gamma[c0 + 1] + beta[c0 + 1];
    unsigned lo = f2bfu(silu(v0));
    unsigned hi = f2bfu(silu(v1));
    dst[i] = lo | (hi << 16);
}

// ---------------- emb_out = silu(emb) @ emb_w + emb_b ----------------------
__global__ void k_emb(const float* __restrict__ emb, const float* __restrict__ ew,
                      const float* __restrict__ eb, float* __restrict__ out) {
    int i = blockIdx.x * blockDim.x + threadIdx.x;
    if (i >= Bb * COUT) return;
    int b = i / COUT, c = i % COUT;
    float acc = eb[c];
    for (int k = 0; k < EMBc; ++k)
        acc += silu(emb[b * EMBc + k]) * ew[k * COUT + c];
    out[i] = acc;
}

// ------- weights [497,64] f32 -> transposed padded bf16 [64][512] ----------
__global__ void k_wconv(const float* __restrict__ w, unsigned short* __restrict__ wbT) {
    int i = blockIdx.x * blockDim.x + threadIdx.x;
    if (i >= COUT * KK) return;
    int ncol = i / KK, k = i % KK;
    int t = k / NCP, c = k % NCP;
    float v = (t < NET && c < NC) ? w[(t * NC + c) * COUT + ncol] : 0.f;
    wbT[(size_t)ncol * KK + k] = f2bfu(v);
}

// -------- edge scatter, packed-bf16 atomics into bf16 agg ------------------
// src: [N][32] packed bf16 pairs; agg: bf16 [N][512]; 32 threads per edge.
__global__ void k_scatter(const unsigned* __restrict__ src,
                          const int* __restrict__ erow, const int* __restrict__ ecol,
                          const int* __restrict__ etyp, const int* __restrict__ ntyp,
                          unsigned short* __restrict__ agg) {
    int tid = blockIdx.x * blockDim.x + threadIdx.x;
    int e  = tid >> 5;           // 32 threads per edge
    int cp = tid & 31;
    if (e >= Ee) return;
    int row = erow[e], col = ecol[e], t = etyp[e];
    size_t base = (size_t)row * KK + t * NCP;
    unsigned payload = src[(size_t)col * (Cc / 2) + cp];
    atomPkAddBf16(agg + base + 2 * cp, payload);
    if (cp == 0) {
        int pos = Cc + ntyp[col];                      // 64..70
        unsigned oh = (pos & 1) ? 0x3F800000u : 0x00003F80u;  // 1.0bf16 in hi/lo half
        atomPkAddBf16(agg + base + (pos & ~1), oh);
    }
}

// ---------------- WMMA GEMM: bf16 [N,512] @ wbT bf16 -> f32 [N,64] ---------
// B (64KB) staged once per workgroup into LDS in fragment-major layout;
// each wave owns 2 M-tiles (32 rows) x full N (4 tiles); A fragments are
// direct 2x16B bf16 loads (no conversion), double-buffered across K.
// mode 0: dst = A@W + extra[batch_id[r]][c]   (embedding add)
// mode 1: dst = A@W + extra[r][c]             (residual add)
__global__ void __launch_bounds__(256)
k_gemm(const unsigned short* __restrict__ agg, const unsigned short* __restrict__ wbT,
       const float* __restrict__ extra, const int* __restrict__ bid,
       int mode, float* __restrict__ dst) {
    __shared__ unsigned char ldsB[64 * 1024];   // 64 frags * 32 lanes * 32B

    // ---- cooperative B preload: frag-major so a lane's frag is 32B contiguous
    for (int it = 0; it < 8; ++it) {
        int idx = threadIdx.x + 256 * it;       // 2048 chunks of 32B
        int f = idx >> 5, l = idx & 31;
        int kt = f >> 2, nt = f & 3;
        int ncol  = nt * 16 + (l & 15);
        int kbase = kt * 32 + 8 * (l >> 4);
        const uint4* gp = (const uint4*)(wbT + (size_t)ncol * KK + kbase);
        uint4 a0 = gp[0], a1 = gp[2];
        uint4* sp = (uint4*)ldsB + (size_t)idx * 2;
        sp[0] = a0; sp[1] = a1;
    }
    __syncthreads();

    int lane  = threadIdx.x & 31;
    int wave  = threadIdx.x >> 5;
    int gwave = blockIdx.x * 8 + wave;           // 8 waves per block
    int m0    = gwave * 32;                      // 32 rows (2 M-tiles) per wave
    int half  = lane >> 4;
    int lm    = lane & 15;

    const unsigned short* arow[2];
    arow[0] = agg + (size_t)(m0 + lm) * KK + 8 * half;
    arow[1] = arow[0] + (size_t)16 * KK;

    v8f acc[2][4];
    for (int mt = 0; mt < 2; ++mt)
        for (int nt = 0; nt < 4; ++nt)
            for (int i = 0; i < 8; ++i) acc[mt][nt][i] = 0.f;

    union AFrag { v16bf v; uint4 q[2]; };
    AFrag cur[2], nxt[2];
#pragma unroll
    for (int mt = 0; mt < 2; ++mt) {
        cur[mt].q[0] = *(const uint4*)(arow[mt]);
        cur[mt].q[1] = *(const uint4*)(arow[mt] + 16);
    }

    const v16bf* lB = (const v16bf*)ldsB;

    for (int kt = 0; kt < KK / 32; ++kt) {
        if (kt + 1 < KK / 32) {
#pragma unroll
            for (int mt = 0; mt < 2; ++mt) {
                const unsigned short* ab = arow[mt] + (kt + 1) * 32;
                nxt[mt].q[0] = *(const uint4*)(ab);
                nxt[mt].q[1] = *(const uint4*)(ab + 16);
            }
        }
#pragma unroll
        for (int nt = 0; nt < 4; ++nt) {
            v16bf bfrag = lB[(kt * 4 + nt) * 32 + lane];
            acc[0][nt] = __builtin_amdgcn_wmma_f32_16x16x32_bf16(
                false, cur[0].v, false, bfrag, (short)0, acc[0][nt], false, false);
            acc[1][nt] = __builtin_amdgcn_wmma_f32_16x16x32_bf16(
                false, cur[1].v, false, bfrag, (short)0, acc[1][nt], false, false);
        }
        cur[0] = nxt[0];
        cur[1] = nxt[1];
    }

    // C/D layout: VGPR i -> M = i + 8*half, N = lane&15
#pragma unroll
    for (int mt = 0; mt < 2; ++mt) {
        for (int i = 0; i < 8; ++i) {
            int r = m0 + mt * 16 + i + 8 * half;
            const float* ex = (mode == 0) ? (extra + (size_t)bid[r] * COUT)
                                          : (extra + (size_t)r * COUT);
#pragma unroll
            for (int nt = 0; nt < 4; ++nt) {
                int c = nt * 16 + lm;
                dst[(size_t)r * COUT + c] = acc[mt][nt][i] + ex[c];
            }
        }
    }
}

extern "C" void kernel_launch(void* const* d_in, const int* in_sizes, int n_in,
                              void* d_out, int out_size, void* d_ws, size_t ws_size,
                              hipStream_t stream) {
    const float* x      = (const float*)d_in[0];
    const float* emb    = (const float*)d_in[1];
    const float* w1     = (const float*)d_in[2];
    const float* w2     = (const float*)d_in[3];
    const float* gamma1 = (const float*)d_in[4];
    const float* beta1  = (const float*)d_in[5];
    const float* gamma2 = (const float*)d_in[6];
    const float* beta2  = (const float*)d_in[7];
    const float* emb_w  = (const float*)d_in[8];
    const float* emb_b  = (const float*)d_in[9];
    const int* erow = (const int*)d_in[10];
    const int* ecol = (const int*)d_in[11];
    const int* etyp = (const int*)d_in[12];
    const int* ntyp = (const int*)d_in[13];
    const int* bid  = (const int*)d_in[14];
    float* out = (float*)d_out;

    // workspace carve-up (256B aligned)
    char* base = (char*)d_ws;
    size_t o = 0;
    auto alloc = [&](size_t bytes) { size_t r = o; o = (o + bytes + 255) & ~(size_t)255; return r; };
    unsigned short* agg = (unsigned short*)(base + alloc((size_t)Nn * KK * 2));   // 134 MB bf16
    unsigned* h1  = (unsigned*)(base + alloc((size_t)Nn * (Cc / 2) * 4));         // packed bf16
    unsigned* h2  = (unsigned*)(base + alloc((size_t)Nn * (Cc / 2) * 4));         // packed bf16
    float* hc     = (float*)(base + alloc((size_t)Nn * COUT * 4));
    float* embv   = (float*)(base + alloc(Bb * COUT * 4));
    float* sums1  = (float*)(base + alloc(Bb * Cc * 4));
    float* sumsq1 = (float*)(base + alloc(Bb * Cc * 4));
    float* cnt1   = (float*)(base + alloc(Bb * 4));
    float* sums2  = (float*)(base + alloc(Bb * Cc * 4));
    float* sumsq2 = (float*)(base + alloc(Bb * Cc * 4));
    float* cnt2   = (float*)(base + alloc(Bb * 4));
    unsigned short* wbT1 = (unsigned short*)(base + alloc((size_t)COUT * KK * 2));
    unsigned short* wbT2 = (unsigned short*)(base + alloc((size_t)COUT * KK * 2));
    (void)ws_size; (void)n_in; (void)in_sizes; (void)out_size;

    // zero stat accumulators
    hipMemsetAsync(sums1, 0, Bb * Cc * 4, stream);
    hipMemsetAsync(sumsq1, 0, Bb * Cc * 4, stream);
    hipMemsetAsync(cnt1, 0, Bb * 4, stream);
    hipMemsetAsync(sums2, 0, Bb * Cc * 4, stream);
    hipMemsetAsync(sumsq2, 0, Bb * Cc * 4, stream);
    hipMemsetAsync(cnt2, 0, Bb * 4, stream);

    // weight relayout + embedding GEMM (independent of pipeline)
    k_wconv<<<(COUT * KK + 255) / 256, 256, 0, stream>>>(w1, wbT1);
    k_wconv<<<(COUT * KK + 255) / 256, 256, 0, stream>>>(w2, wbT2);
    k_emb<<<(Bb * COUT + 255) / 256, 256, 0, stream>>>(emb, emb_w, emb_b, embv);

    // --- GN1 + SiLU (writes packed bf16 h1) ---
    const int RPB = 128;
    k_stats<<<Nn / RPB, 64, 0, stream>>>(x, bid, sums1, sumsq1, cnt1, RPB);
    k_gn_silu<<<(Nn * (Cc / 2)) / 256, 256, 0, stream>>>(
        x, bid, sums1, sumsq1, cnt1, gamma1, beta1, h1);

    // --- GraphConv1 + emb add ---
    hipMemsetAsync(agg, 0, (size_t)Nn * KK * 2, stream);
    k_scatter<<<(Ee * 32) / 256, 256, 0, stream>>>(h1, erow, ecol, etyp, ntyp, agg);
    k_gemm<<<Nn / 32 / 8, 256, 0, stream>>>(agg, wbT1, embv, bid, 0, hc);

    // --- GN2 + SiLU (writes packed bf16 h2) ---
    k_stats<<<Nn / RPB, 64, 0, stream>>>(hc, bid, sums2, sumsq2, cnt2, RPB);
    k_gn_silu<<<(Nn * (COUT / 2)) / 256, 256, 0, stream>>>(
        hc, bid, sums2, sumsq2, cnt2, gamma2, beta2, h2);

    // --- GraphConv2 + residual ---
    hipMemsetAsync(agg, 0, (size_t)Nn * KK * 2, stream);
    k_scatter<<<(Ee * 32) / 256, 256, 0, stream>>>(h2, erow, ecol, etyp, ntyp, agg);
    k_gemm<<<Nn / 32 / 8, 256, 0, stream>>>(agg, wbT2, x, bid, 1, out);
}
